// GCN_56341380989304
// MI455X (gfx1250) — matrix operand order repaired and checked by reading, not compiled, not run on hardware
//
#include <hip/hip_runtime.h>
#include <hip/hip_bf16.h>
#include <stdint.h>

// ---- problem sizes (fixed by the reference) ----
#define NN 50000    // nodes
#define NE 800000   // edges (self-loops added on top)
#define NF 128      // feature width (= HIDDEN)
#define NG 64       // graphs

typedef __attribute__((ext_vector_type(16))) _Float16 v16h;
typedef __attribute__((ext_vector_type(8)))  float    v8f;

// ---------------------------------------------------------------- utilities
__global__ void fill_kernel(float* __restrict__ p, float v, int n) {
    int i = blockIdx.x * blockDim.x + threadIdx.x;
    if (i < n) p[i] = v;
}

__global__ void deg_accum_kernel(float* __restrict__ deg,
                                 const int* __restrict__ dst, int ne) {
    int i = blockIdx.x * blockDim.x + threadIdx.x;
    if (i < ne) atomicAdd(&deg[dst[i]], 1.0f);
}

__global__ void rsqrt_kernel(float* __restrict__ deg, int n) {
    int i = blockIdx.x * blockDim.x + threadIdx.x;
    if (i < n) deg[i] = rsqrtf(deg[i]);   // deg >= 1 always (self-loop)
}

// Wt[n*128+k] = (f16) W[k*128+n]  -- transpose so WMMA B-frag reads are contiguous
__global__ void prep_wt_kernel(const float* __restrict__ W,
                               _Float16* __restrict__ Wt) {
    int t = blockIdx.x * blockDim.x + threadIdx.x;
    if (t < NF * NF) {
        int n = t >> 7, k = t & 127;
        Wt[t] = (_Float16)W[k * NF + n];
    }
}

// ------------------------------------------------------- WMMA dense GEMM
// Out[NNrows x 128] = X[NNrows x 128] @ W[128 x 128]
// One block = 16 output rows x 128 cols = 8 waves, each wave one 16x16 tile,
// K traversed in 4 steps of 32 with v_wmma_f32_16x16x32_f16.
__global__ __launch_bounds__(256)
void gemm_wmma_kernel(const float* __restrict__ X,
                      const _Float16* __restrict__ Wt,   // [N][K] f16 (transposed)
                      float* __restrict__ Out) {
    __shared__ _Float16 sWt[NF * NF];   // 32 KB
    __shared__ _Float16 sX[16 * NF];    //  4 KB

    const int tid = threadIdx.x;
    const int m0  = blockIdx.x * 16;

    // cooperative stage of transposed f16 weights (dword copies)
    {
        const uint32_t* w32 = (const uint32_t*)Wt;
        uint32_t*       s32 = (uint32_t*)sWt;
        for (int i = tid; i < (NF * NF) / 2; i += 256) s32[i] = w32[i];
    }
    // stage + convert this block's 16 input rows
    for (int i = tid; i < 16 * NF; i += 256)
        sX[i] = (_Float16)X[(size_t)m0 * NF + i];
    __syncthreads();

    const int lane = tid & 31;
    const int wv   = tid >> 5;
    const int hi   = lane >> 4;     // K half-block selector
    const int lo   = lane & 15;     // M (A) / N (B,C,D) index
    const int n0   = wv * 16;

    v8f c = {};
    #pragma unroll
    for (int k0 = 0; k0 < NF; k0 += 32) {
        v16h a, b;
        #pragma unroll
        for (int e = 0; e < 16; ++e) {
            // A 16x32 f16 layout: lanes hold M=lo; VGPR halves map to
            // K = k0 + (e/8)*16 + hi*8 + (e%8)
            a[e] = sX[lo * NF + k0 + ((e >> 3) << 4) + (hi << 3) + (e & 7)];
            // B 32x16 f16 layout: lanes hold N=lo; K = k0 + (e/8)*16 + hi*8 + (e%8)
            b[e] = sWt[(n0 + lo) * NF + k0 + ((e >> 3) << 4) + (hi << 3) + (e & 7)];
        }
        c = __builtin_amdgcn_wmma_f32_16x16x32_f16(
                /*neg_a=*/false, a, /*neg_b=*/false, b,
                /*c_mod=*/(short)0, c, /*reuse_a=*/false, /*reuse_b=*/false);
    }
    // C/D 16x16 f32 layout: VGPR r -> row m0 + r + 8*hi, col n0 + lo
    #pragma unroll
    for (int r = 0; r < 8; ++r)
        Out[(size_t)(m0 + r + 8 * hi) * NF + n0 + lo] = c[r];
}

// ------------------------------------------------- edge scatter (roofline path)
// One wave per logical edge (NE real edges + NN self-loops).
// Each lane moves 4 features: float4 gather + 4 f32 atomic adds.
__global__ void scatter_kernel(const float* __restrict__ A,
                               float* __restrict__ B,
                               const int* __restrict__ src,
                               const int* __restrict__ dst,
                               const float* __restrict__ dinv) {
    int gtid = blockIdx.x * blockDim.x + threadIdx.x;
    int wave = gtid >> 5;
    int lane = gtid & 31;
    if (wave >= NE + NN) return;

    int s, d;
    if (wave < NE) { s = src[wave]; d = dst[wave]; }
    else           { s = d = wave - NE; }          // self-loop

    const float* arow = A + (size_t)s * NF;
    __builtin_prefetch(arow + lane * 4, 0, 0);     // global_prefetch_b8

    float norm = dinv[s] * dinv[d];
    float4 v = ((const float4*)arow)[lane];
    float* brow = B + (size_t)d * NF + lane * 4;
    atomicAdd(brow + 0, v.x * norm);
    atomicAdd(brow + 1, v.y * norm);
    atomicAdd(brow + 2, v.z * norm);
    atomicAdd(brow + 3, v.w * norm);
}

// ------------------------------------------------- bias + activation (fused)
template <int ACT>  // 0 = sigmoid, 1 = relu
__global__ void bias_act_kernel(float* __restrict__ B,
                                const float* __restrict__ bias, int total) {
    int i = blockIdx.x * blockDim.x + threadIdx.x;
    if (i >= total) return;
    float v = B[i] + bias[i & (NF - 1)];
    if (ACT == 0) v = 1.0f / (1.0f + __expf(-v));
    else          v = fmaxf(v, 0.0f);
    B[i] = v;
}

// ------------------------------------------------- global mean pool (atomics)
__global__ void pool_kernel(const float* __restrict__ H,
                            const int* __restrict__ batch,
                            float* __restrict__ pooled,
                            float* __restrict__ cnt) {
    int i = blockIdx.x * blockDim.x + threadIdx.x;
    if (i >= NN * NF) return;
    int n = i >> 7, f = i & (NF - 1);
    int g = batch[n];
    atomicAdd(&pooled[g * NF + f], H[i]);
    if (f == 0) atomicAdd(&cnt[g], 1.0f);
}

// ------------------------------------------------- MLP head, one block/graph
__global__ __launch_bounds__(128)
void mlp_kernel(const float* __restrict__ pooled, const float* __restrict__ cnt,
                const float* __restrict__ ilW, const float* __restrict__ ilb,
                const float* __restrict__ h1W, const float* __restrict__ h1b,
                const float* __restrict__ olW, const float* __restrict__ olb,
                float* __restrict__ out) {
    __shared__ float p[NF];
    __shared__ float s1[64];
    __shared__ float s2[16];
    int g = blockIdx.x, t = threadIdx.x;

    float c = fmaxf(cnt[g], 1.0f);
    p[t] = pooled[g * NF + t] / c;
    __syncthreads();

    if (t < 64) {
        float acc = ilb[t];
        #pragma unroll 8
        for (int f = 0; f < NF; ++f) acc += p[f] * ilW[f * 64 + t];
        s1[t] = 1.0f / (1.0f + __expf(-acc));
    }
    __syncthreads();
    if (t < 16) {
        float acc = h1b[t];
        #pragma unroll 8
        for (int i = 0; i < 64; ++i) acc += s1[i] * h1W[i * 16 + t];
        s2[t] = fmaxf(acc, 0.0f);
    }
    __syncthreads();
    if (t == 0) {
        float acc = olb[0];
        #pragma unroll
        for (int i = 0; i < 16; ++i) acc += s2[i] * olW[i];
        out[g] = acc;
    }
}

// ------------------------------------------------------------------ launcher
extern "C" void kernel_launch(void* const* d_in, const int* in_sizes, int n_in,
                              void* d_out, int out_size, void* d_ws, size_t ws_size,
                              hipStream_t stream) {
    const float* x   = (const float*)d_in[0];
    const float* W1  = (const float*)d_in[1];
    const float* b1  = (const float*)d_in[2];
    const float* W2  = (const float*)d_in[3];
    const float* b2  = (const float*)d_in[4];
    const float* ilW = (const float*)d_in[5];
    const float* ilb = (const float*)d_in[6];
    const float* h1W = (const float*)d_in[7];
    const float* h1b = (const float*)d_in[8];
    const float* olW = (const float*)d_in[9];
    const float* olb = (const float*)d_in[10];
    const int* ei    = (const int*)d_in[11];   // harness delivers integer inputs as int32
    const int* batch = (const int*)d_in[12];
    const int* src = ei;
    const int* dst = ei + NE;
    float* out = (float*)d_out;                // [64]

    // workspace carve-up (floats), all 16B-aligned slices
    float*     ws     = (float*)d_ws;
    float*     deg    = ws;                                  // 50000 (-> dinv)
    float*     A      = ws + 50176;                          // 6.4M  (xW / h1W2)
    float*     Bb     = A + (size_t)NN * NF;                 // 6.4M  (scatter tgt / h)
    float*     pooled = Bb + (size_t)NN * NF;                // 64*128
    float*     cnt    = pooled + NG * NF;                    // 64 (padded 128)
    _Float16*  Wt     = (_Float16*)(cnt + 128);              // 128*128 f16

    const int TPB = 256;
    const int gNodes   = (NN + TPB - 1) / TPB;               // 196
    const int gEdges   = (NE + TPB - 1) / TPB;               // 3125
    const int gFeat    = (NN * NF) / TPB;                    // 25000
    const int gScatter = ((NE + NN) * 32 + TPB - 1) / TPB;   // 106250
    const int gGemm    = NN / 16;                            // 3125

    // degrees (self-loop folded in via init=1) -> dinv
    fill_kernel<<<gNodes, TPB, 0, stream>>>(deg, 1.0f, NN);
    fill_kernel<<<(NG * NF + 64 + TPB - 1) / TPB, TPB, 0, stream>>>(pooled, 0.0f, NG * NF + 64);
    deg_accum_kernel<<<gEdges, TPB, 0, stream>>>(deg, dst, NE);
    rsqrt_kernel<<<gNodes, TPB, 0, stream>>>(deg, NN);

    // ---- layer 1: A = x @ W1 ; B = sigmoid(scatter(A) + b1)
    prep_wt_kernel<<<(NF * NF) / TPB, TPB, 0, stream>>>(W1, Wt);
    gemm_wmma_kernel<<<gGemm, TPB, 0, stream>>>(x, Wt, A);
    fill_kernel<<<gFeat, TPB, 0, stream>>>(Bb, 0.0f, NN * NF);
    scatter_kernel<<<gScatter, TPB, 0, stream>>>(A, Bb, src, dst, deg);
    bias_act_kernel<0><<<gFeat, TPB, 0, stream>>>(Bb, b1, NN * NF);

    // ---- layer 2: A = B @ W2 ; B = relu(scatter(A) + b2)
    prep_wt_kernel<<<(NF * NF) / TPB, TPB, 0, stream>>>(W2, Wt);
    gemm_wmma_kernel<<<gGemm, TPB, 0, stream>>>(Bb, Wt, A);
    fill_kernel<<<gFeat, TPB, 0, stream>>>(Bb, 0.0f, NN * NF);
    scatter_kernel<<<gScatter, TPB, 0, stream>>>(A, Bb, src, dst, deg);
    bias_act_kernel<1><<<gFeat, TPB, 0, stream>>>(Bb, b2, NN * NF);

    // ---- mean pool + MLP head
    pool_kernel<<<gFeat, TPB, 0, stream>>>(Bb, batch, pooled, cnt);
    mlp_kernel<<<NG, 128, 0, stream>>>(pooled, cnt, ilW, ilb, h1W, h1b, olW, olb, out);
}
